// ErecRAM_55233279426702
// MI455X (gfx1250) — matrix operand order repaired and checked by reading, not compiled, not run on hardware
//
#include <hip/hip_runtime.h>
#include <hip/hip_bf16.h>
#include <math.h>

typedef __attribute__((ext_vector_type(16))) _Float16 v16h;
typedef __attribute__((ext_vector_type(8)))  float    v8f;

#define STATE_DIM 4096
#define INV_SQRT_D (1.0f / 64.0f)
#define LAMBDA_DECAY 0.01f
#define ALPHA_BLEND 0.95f
#define LN_EPS 1e-5f

// ---------------------------------------------------------------------------
// K0: init workspace (max = -inf, sum_p = 0, acc[4096] = 0)
// ---------------------------------------------------------------------------
__global__ void k_init(float* __restrict__ maxv, float* __restrict__ sump,
                       float* __restrict__ acc) {
    int i = blockIdx.x * blockDim.x + threadIdx.x;
    if (i < STATE_DIM) acc[i] = 0.0f;
    if (i == 0) { maxv[0] = -INFINITY; sump[0] = 0.0f; }
}

// ---------------------------------------------------------------------------
// K1: logits[m] = (states[m] . q) / 64 * w[m] * exp(-0.01*|t_new - ts[m]|)
// One wave computes 16 rows via v_wmma_f32_16x16x32_f16, accumulating over
// K = 4096 in chunks of 32. Query is broadcast into all 16 B columns, so
// every column of D equals the 16-row dot-product vector.
// ---------------------------------------------------------------------------
__global__ void __launch_bounds__(256)
k_logits(const float* __restrict__ q, const float* __restrict__ states,
         const float* __restrict__ ts, const float* __restrict__ w,
         const float* __restrict__ t_new, float* __restrict__ logits, int M) {
    const int lane   = threadIdx.x & 31;
    const int waveId = (blockIdx.x * (blockDim.x >> 5)) + (threadIdx.x >> 5);
    const int rowBase = waveId * 16;
    if (rowBase >= M) return;  // wave-uniform exit

    // A-matrix (16x32 f16) per-lane layout (ISA 7.12.2):
    //   lanes 0-15 : M = lane,    h[0..7] = K 0..7,   h[8..15] = K 16..23
    //   lanes 16-31: M = lane-16, h[0..7] = K 8..15,  h[8..15] = K 24..31
    const int m      = lane & 15;
    int row = rowBase + m;
    if (row >= M) row = M - 1;                 // clamp, masked on writeback
    const float* rowPtr = states + (size_t)row * STATE_DIM;
    const int kHalfA = (lane < 16) ? 0 : 8;
    // B-matrix (32x16 f16): same K order as A (lanes 0-15 K 0..7,16..23; lanes 16-31 K 8..15,24..31).
    const int kHalfB = (lane < 16) ? 0 : 8;

    v8f c = {};
    #pragma unroll 2
    for (int kc = 0; kc < STATE_DIM; kc += 32) {
        const float* ap0 = rowPtr + kc + kHalfA;        // 8 contiguous f32
        const float* ap1 = ap0 + 16;                    // 8 contiguous f32
        const float* qp  = q + kc + kHalfB;             // 8 + 8 f32 (offset 16)
        v16h a, b;
        #pragma unroll
        for (int j = 0; j < 8; ++j) {
            a[j]     = (_Float16)ap0[j];
            a[8 + j] = (_Float16)ap1[j];
        }
        #pragma unroll
        for (int j = 0; j < 8; ++j) { b[j] = (_Float16)qp[j]; b[8 + j] = (_Float16)qp[16 + j]; }
        c = __builtin_amdgcn_wmma_f32_16x16x32_f16(
                /*neg_a=*/false, a, /*neg_b=*/false, b,
                /*c_mod=*/(short)0, c, /*reuse_a=*/false, /*reuse_b=*/false);
    }

    // C/D layout: VGPR r -> M=r (lanes 0-15), M=r+8 (lanes 16-31); all N equal.
    if (lane == 0 || lane == 16) {
        const float tnew = t_new[0];
        const int base = rowBase + ((lane == 16) ? 8 : 0);
        #pragma unroll
        for (int r = 0; r < 8; ++r) {
            const int rr = base + r;
            if (rr < M) {
                const float dt = tnew - ts[rr];
                const float dw = w[rr] * __expf(-LAMBDA_DECAY * fabsf(dt));
                logits[rr] = c[r] * INV_SQRT_D * dw;
            }
        }
    }
}

// ---------------------------------------------------------------------------
// K2: global max of logits (single block)
// ---------------------------------------------------------------------------
__global__ void k_max(const float* __restrict__ logits, float* __restrict__ maxv,
                      int M) {
    __shared__ float red[1024];
    float mx = -INFINITY;
    for (int i = threadIdx.x; i < M; i += blockDim.x) mx = fmaxf(mx, logits[i]);
    red[threadIdx.x] = mx;
    __syncthreads();
    for (int s = blockDim.x >> 1; s > 0; s >>= 1) {
        if (threadIdx.x < (unsigned)s)
            red[threadIdx.x] = fmaxf(red[threadIdx.x], red[threadIdx.x + s]);
        __syncthreads();
    }
    if (threadIdx.x == 0) maxv[0] = red[0];
}

// ---------------------------------------------------------------------------
// K3: p[m] = exp(logit[m]-max); acc[d] += p[m]*states[m][d]; sum_p += p[m]
// 256 threads/block, thread t owns columns [16t, 16t+16) -> fully coalesced.
// ---------------------------------------------------------------------------
__global__ void __launch_bounds__(256)
k_accumulate(const float* __restrict__ states, const float* __restrict__ logits,
             const float* __restrict__ maxv, float* __restrict__ acc,
             float* __restrict__ sump, int M) {
    const int t  = threadIdx.x;
    const int c0 = t * 16;
    const float mx = maxv[0];
    float av[16];
    #pragma unroll
    for (int i = 0; i < 16; ++i) av[i] = 0.0f;
    float localSum = 0.0f;

    for (int mrow = blockIdx.x; mrow < M; mrow += gridDim.x) {
        const float p = __expf(logits[mrow] - mx);
        if (t == 0) localSum += p;
        const float4* sp =
            reinterpret_cast<const float4*>(states + (size_t)mrow * STATE_DIM + c0);
        #pragma unroll
        for (int v = 0; v < 4; ++v) {
            const float4 s4 = sp[v];
            av[4 * v + 0] = fmaf(p, s4.x, av[4 * v + 0]);
            av[4 * v + 1] = fmaf(p, s4.y, av[4 * v + 1]);
            av[4 * v + 2] = fmaf(p, s4.z, av[4 * v + 2]);
            av[4 * v + 3] = fmaf(p, s4.w, av[4 * v + 3]);
        }
    }
    #pragma unroll
    for (int i = 0; i < 16; ++i) atomicAdd(&acc[c0 + i], av[i]);
    if (t == 0) atomicAdd(sump, localSum);
}

// ---------------------------------------------------------------------------
// K4: out = LayerNorm(0.95*q + 0.05*acc/sum_p)   (single block, 256 threads)
// ---------------------------------------------------------------------------
__global__ void __launch_bounds__(256)
k_finalize(const float* __restrict__ q, const float* __restrict__ acc,
           const float* __restrict__ sump, float* __restrict__ out) {
    __shared__ float s1[256];
    __shared__ float s2[256];
    const int t  = threadIdx.x;
    const int c0 = t * 16;
    const float inv = 1.0f / sump[0];
    float v[16];
    float ls = 0.0f, lq = 0.0f;
    #pragma unroll
    for (int i = 0; i < 16; ++i) {
        const float nv = ALPHA_BLEND * q[c0 + i] + (1.0f - ALPHA_BLEND) * acc[c0 + i] * inv;
        v[i] = nv;
        ls += nv;
        lq += nv * nv;
    }
    s1[t] = ls; s2[t] = lq;
    __syncthreads();
    for (int s = 128; s > 0; s >>= 1) {
        if (t < s) { s1[t] += s1[t + s]; s2[t] += s2[t + s]; }
        __syncthreads();
    }
    const float mu  = s1[0] * (1.0f / STATE_DIM);
    const float var = s2[0] * (1.0f / STATE_DIM) - mu * mu;
    const float r   = rsqrtf(var + LN_EPS);
    #pragma unroll
    for (int i = 0; i < 16; ++i) out[c0 + i] = (v[i] - mu) * r;
}

// ---------------------------------------------------------------------------
extern "C" void kernel_launch(void* const* d_in, const int* in_sizes, int n_in,
                              void* d_out, int out_size, void* d_ws, size_t ws_size,
                              hipStream_t stream) {
    const float* q      = (const float*)d_in[0];  // current_state [4096]
    const float* states = (const float*)d_in[1];  // [M, 4096]
    const float* ts     = (const float*)d_in[2];  // [M]
    const float* w      = (const float*)d_in[3];  // [M]
    const float* t_new  = (const float*)d_in[4];  // [1]
    float* out = (float*)d_out;
    const int M = in_sizes[2];

    // workspace layout (floats): [0]=max, [1]=sum_p, [4..4+4096)=acc, then logits[M]
    float* wsf    = (float*)d_ws;
    float* maxv   = wsf;
    float* sump   = wsf + 1;
    float* acc    = wsf + 4;
    float* logits = wsf + 4 + STATE_DIM;

    k_init<<<(STATE_DIM + 255) / 256, 256, 0, stream>>>(maxv, sump, acc);

    const int waves   = (M + 15) / 16;      // 16 rows per wave
    const int blocks1 = (waves + 7) / 8;    // 8 waves (256 threads) per block
    k_logits<<<blocks1, 256, 0, stream>>>(q, states, ts, w, t_new, logits, M);

    k_max<<<1, 1024, 0, stream>>>(logits, maxv, M);

    k_accumulate<<<512, 256, 0, stream>>>(states, logits, maxv, acc, sump, M);

    k_finalize<<<1, 256, 0, stream>>>(q, acc, sump, out);
}
